// TritonConv2dINT8_85461259256098
// MI455X (gfx1250) — 8-bit matrix operand order repaired and checked by reading, compiled, not run on hardware
//
#include <hip/hip_runtime.h>


typedef __attribute__((ext_vector_type(4))) int v4i;
typedef __attribute__((ext_vector_type(8))) int v8i;

#define B_   32
#define C_   256
#define H_   56
#define W_   56
#define HP_  58
#define WP_  58
#define HW_  (H_ * W_)
#define MPIX (B_ * HW_)
#define KTAP 9
#define WTAP (C_ * C_)

#define WS_ACT_OFF  256
#define WS_ACT_SIZE (B_ * HP_ * WP_ * C_)
#define WS_WGT_OFF  (WS_ACT_OFF + WS_ACT_SIZE)

__global__ __launch_bounds__(64) void init_kernel(unsigned* amax) {
    if (threadIdx.x == 0) amax[0] = 0u;
}

__global__ __launch_bounds__(256) void absmax_kernel(const float* __restrict__ x,
                                                     unsigned* __restrict__ amax,
                                                     int n) {
    __shared__ unsigned red[256];
    unsigned m = 0u;
    for (int i = blockIdx.x * 256 + threadIdx.x; i < n; i += gridDim.x * 256)
        m = max(m, __float_as_uint(fabsf(x[i])));
    red[threadIdx.x] = m;
    __syncthreads();
    for (int s = 128; s > 0; s >>= 1) {
        if (threadIdx.x < s)
            red[threadIdx.x] = max(red[threadIdx.x], red[threadIdx.x + s]);
        __syncthreads();
    }
    if (threadIdx.x == 0) atomicMax(amax, red[0]);
}

__global__ __launch_bounds__(256) void quant_kernel(const float* __restrict__ x,
                                                    const unsigned* __restrict__ amax_bits,
                                                    signed char* __restrict__ act) {
    int idx  = blockIdx.x * 256 + threadIdx.x;
    int cin4 = idx & 63;
    int t    = idx >> 6;
    int iwp  = t % WP_;
    int t2   = t / WP_;
    int ihp  = t2 % HP_;
    int b    = t2 / HP_;

    int cin0 = cin4 * 4;
    int k    = cin0 & 63;
    int g    = cin0 >> 6;
    int pos  = g * 64 + ((k >> 4) & 1) * 32 + ((k >> 5) & 1) * 16 + (k & 15);

    unsigned val = 0u;
    if (ihp >= 1 && ihp <= H_ && iwp >= 1 && iwp <= W_) {
        float inv = 127.0f / __uint_as_float(amax_bits[0]);
        int ih = ihp - 1, iw = iwp - 1;
        const float* xp = x + (((b * C_ + cin0) * H_ + ih) * W_ + iw);
#pragma unroll
        for (int j = 0; j < 4; ++j) {
            float q = rintf(xp[j * HW_] * inv);
            q = fminf(127.0f, fmaxf(-127.0f, q));
            val |= ((unsigned)((int)q & 0xff)) << (8 * j);
        }
    }
    *reinterpret_cast<unsigned*>(act + ((b * HP_ + ihp) * WP_ + iwp) * C_ + pos) = val;
}

__global__ __launch_bounds__(256) void packw_kernel(const int* __restrict__ wq,
                                                    signed char* __restrict__ wgt) {
    int idx  = blockIdx.x * 256 + threadIdx.x;
    int cin4 = idx & 63;
    int cout = (idx >> 6) & 255;
    int tap  = idx >> 14;
    int cin0 = cin4 * 4;

    const int* src = wq + (cout * C_ + cin0) * KTAP + tap;
    unsigned val = 0u;
#pragma unroll
    for (int j = 0; j < 4; ++j)
        val |= ((unsigned)(src[j * KTAP] & 0xff)) << (8 * j);
    *reinterpret_cast<unsigned*>(wgt + tap * WTAP + cout * C_ + cin0) = val;
}

__device__ __forceinline__ v8i load_frag(const signed char* p, int delta) {
    v4i lo = *reinterpret_cast<const v4i*>(p);
    v4i hi = *reinterpret_cast<const v4i*>(p + delta);
    return __builtin_shufflevector(lo, hi, 0, 1, 2, 3, 4, 5, 6, 7);
}

__global__ __launch_bounds__(256) void conv_kernel(const signed char* __restrict__ act,
                                                   const signed char* __restrict__ wgt,
                                                   const unsigned* __restrict__ amax_bits,
                                                   const float* __restrict__ wscale,
                                                   const float* __restrict__ bias,
                                                   float* __restrict__ out) {
    const int lane  = threadIdx.x & 31;
    const int wave  = threadIdx.x >> 5;
    const int n     = lane & 15;
    const int hh    = lane >> 4;
    const int mbase = blockIdx.x * 32;
    const int coutB = wave * 32;

    unsigned pixA0 = (unsigned)(mbase + n);
    unsigned pixA1 = pixA0 + 16u;
    unsigned bA0 = pixA0 / (unsigned)HW_, rA0 = pixA0 % (unsigned)HW_;
    unsigned bA1 = pixA1 / (unsigned)HW_, rA1 = pixA1 % (unsigned)HW_;
    unsigned ohA0 = rA0 / (unsigned)W_, owA0 = rA0 % (unsigned)W_;
    unsigned ohA1 = rA1 / (unsigned)W_, owA1 = rA1 % (unsigned)W_;
    const signed char* aBase0 = act + (((bA0 * HP_ + ohA0) * WP_ + owA0) * C_) + hh * 32;
    const signed char* aBase1 = act + (((bA1 * HP_ + ohA1) * WP_ + owA1) * C_) + hh * 32;

    const signed char* b0Base = wgt + (coutB + n) * C_ + hh * 16;
    const signed char* b1Base = b0Base + 16 * C_;

    v8i acc00 = {0, 0, 0, 0, 0, 0, 0, 0};
    v8i acc01 = {0, 0, 0, 0, 0, 0, 0, 0};
    v8i acc10 = {0, 0, 0, 0, 0, 0, 0, 0};
    v8i acc11 = {0, 0, 0, 0, 0, 0, 0, 0};

#pragma unroll
    for (int kh = 0; kh < 3; ++kh) {
#pragma unroll
        for (int kw = 0; kw < 3; ++kw) {
            const signed char* aRow0 = aBase0 + (kh * WP_ + kw) * C_;
            const signed char* aRow1 = aBase1 + (kh * WP_ + kw) * C_;
            const int tapOff = (kh * 3 + kw) * WTAP;
#pragma unroll
            for (int c = 0; c < 4; ++c) {
                v8i a0 = load_frag(aRow0 + c * 64, 16);
                v8i a1 = load_frag(aRow1 + c * 64, 16);
                v8i b0 = load_frag(b0Base + tapOff + c * 64, 32);
                v8i b1 = load_frag(b1Base + tapOff + c * 64, 32);
                acc00 = __builtin_amdgcn_wmma_i32_16x16x64_iu8(
                            true, a0, true, b0, acc00, false, false);
                acc01 = __builtin_amdgcn_wmma_i32_16x16x64_iu8(
                            true, a0, true, b1, acc01, false, false);
                acc10 = __builtin_amdgcn_wmma_i32_16x16x64_iu8(
                            true, a1, true, b0, acc10, false, false);
                acc11 = __builtin_amdgcn_wmma_i32_16x16x64_iu8(
                            true, a1, true, b1, acc11, false, false);
            }
        }
    }

    const float scale = (__uint_as_float(amax_bits[0]) * (1.0f / 127.0f)) * wscale[0];
    const float bi0 = bias[coutB + n];
    const float bi1 = bias[coutB + 16 + n];
#pragma unroll
    for (int r = 0; r < 8; ++r) {
        unsigned pix0 = (unsigned)(mbase + r + hh * 8);
        unsigned pix1 = pix0 + 16u;

        unsigned b0p  = pix0 / (unsigned)HW_, rem0 = pix0 % (unsigned)HW_;
        unsigned oh0  = rem0 / (unsigned)W_,  ow0  = rem0 % (unsigned)W_;
        unsigned o0   = (b0p * C_ + (unsigned)(coutB + n)) * (unsigned)HW_ + oh0 * W_ + ow0;
        out[o0]             = (float)acc00[r] * scale + bi0;
        out[o0 + 16u * HW_] = (float)acc01[r] * scale + bi1;

        unsigned b1p  = pix1 / (unsigned)HW_, rem1 = pix1 % (unsigned)HW_;
        unsigned oh1  = rem1 / (unsigned)W_,  ow1  = rem1 % (unsigned)W_;
        unsigned o1   = (b1p * C_ + (unsigned)(coutB + n)) * (unsigned)HW_ + oh1 * W_ + ow1;
        out[o1]             = (float)acc10[r] * scale + bi0;
        out[o1 + 16u * HW_] = (float)acc11[r] * scale + bi1;
    }
}

extern "C" void kernel_launch(void* const* d_in, const int* in_sizes, int n_in,
                              void* d_out, int out_size, void* d_ws, size_t ws_size,
                              hipStream_t stream) {
    (void)in_sizes; (void)n_in; (void)out_size; (void)ws_size;

    const float* x      = (const float*)d_in[0];
    const int*   wq     = (const int*)d_in[1];
    const float* wscale = (const float*)d_in[2];
    const float* bias   = (const float*)d_in[3];
    float*       out    = (float*)d_out;

    char* ws = (char*)d_ws;
    unsigned*    amax = (unsigned*)ws;
    signed char* act  = (signed char*)(ws + WS_ACT_OFF);
    signed char* wgt  = (signed char*)(ws + WS_WGT_OFF);

    init_kernel<<<1, 64, 0, stream>>>(amax);

    absmax_kernel<<<1024, 256, 0, stream>>>(x, amax, B_ * C_ * H_ * W_);

    const int quantThreads = B_ * HP_ * WP_ * 64;
    quant_kernel<<<quantThreads / 256, 256, 0, stream>>>(x, amax, act);

    packw_kernel<<<(KTAP * C_ * 64) / 256, 256, 0, stream>>>(wq, wgt);

    conv_kernel<<<MPIX / 32, 256, 0, stream>>>(act, wgt, amax, wscale, bias, out);
}
